// SpatialAttention_47321949667959
// MI455X (gfx1250) — matrix operand order repaired and checked by reading, not compiled, not run on hardware
//
#include <hip/hip_runtime.h>
#include <hip/hip_bf16.h>

// Problem sizes (fixed by the reference)
#define B_   8
#define CI   256     // C_IN (also V channels)
#define CO   128     // C_OUT (attention head dim)
#define NN   4096    // H*W

typedef __attribute__((ext_vector_type(16))) __bf16       v16bf;
typedef __attribute__((ext_vector_type(2)))  __bf16       v2bf;
typedef __attribute__((ext_vector_type(8)))  float        v8f;
typedef __attribute__((ext_vector_type(8)))  unsigned int u32x8;
typedef __attribute__((ext_vector_type(4)))  unsigned int u32x4;
typedef __attribute__((ext_vector_type(2)))  unsigned int u32x2;
typedef __attribute__((ext_vector_type(4)))  float        f32x4;

// ---- helpers -------------------------------------------------------------

// Native f32 -> bf16 conversions (clang emits v_cvt_{pk_}bf16_f32)
static __device__ __forceinline__ unsigned short f2bf(float f) {
  return __builtin_bit_cast(unsigned short, (__bf16)f);
}
static __device__ __forceinline__ unsigned int pack2bf(float lo, float hi) {
  v2bf t;
  t[0] = (__bf16)lo;
  t[1] = (__bf16)hi;
  return __builtin_bit_cast(unsigned int, t);
}

// Load a 32-byte WMMA operand fragment as two b128 loads at
// half-word offsets o0 / o1 (works for global and LDS pointers).
static __device__ __forceinline__ v16bf frag_ld(const unsigned short* p, int o0, int o1) {
  u32x4 a = *(const u32x4*)(p + o0);
  u32x4 b = *(const u32x4*)(p + o1);
  u32x8 f;
  f[0] = a[0]; f[1] = a[1]; f[2] = a[2]; f[3] = a[3];
  f[4] = b[0]; f[5] = b[1]; f[6] = b[2]; f[7] = b[3];
  return __builtin_bit_cast(v16bf, f);
}

static __device__ __forceinline__ v8f wmma_bf16(v16bf a, v16bf b, v8f c) {
  // v_wmma_f32_16x16x32_bf16 (8-arg form: neg_a, A, neg_b, B, c_mod, C, reuse_a, reuse_b)
  return __builtin_amdgcn_wmma_f32_16x16x32_bf16(false, a, false, b, (short)0, c, false, false);
}

// Async DMA: global -> LDS, 16 bytes per lane, tracked on ASYNCcnt.
static __device__ __forceinline__ void async_g2l_b128(void* lds, const void* gptr) {
  unsigned laddr = (unsigned)(uintptr_t)lds;   // low 32 bits of generic ptr = LDS offset
  asm volatile("global_load_async_to_lds_b128 %0, %1, off"
               :: "v"(laddr), "v"((unsigned long long)(uintptr_t)gptr)
               : "memory");
}
static __device__ __forceinline__ void wait_async0() {
  asm volatile("s_wait_asynccnt 0x0" ::: "memory");
}

// Tensor Data Mover: 2D tile (tile_d0 bf16 elems x tile_d1 rows, row stride
// ten_d0_stride elems) from global to LDS, with a 4-DWORD LDS pad inserted
// after every 64 DWORDs (=> 136-half LDS pitch for a 128-half row).
// Issued once per wave; tracked on TENSORcnt.
static __device__ __forceinline__ void tdm_load_2d(unsigned lds_addr,
                                                   unsigned long long gaddr,
                                                   unsigned tile_d0, unsigned tile_d1,
                                                   unsigned ten_d0_stride) {
  u32x4 g0;
  g0[0] = 1u;                                   // count=1, user descriptor
  g0[1] = lds_addr;                             // lds_addr [63:32]
  g0[2] = (unsigned)gaddr;                      // global_addr [95:64]
  g0[3] = (unsigned)(gaddr >> 32) | (2u << 30); // global_addr hi | type=2 [127:126]
  u32x8 g1;
  g1[0] = (1u << 16)                            // data_size = 2 bytes
        | (1u << 20)                            // pad_enable
        | (5u << 22)                            // pad_interval: every 64 DWORDs
        | (3u << 25);                           // pad_amount: 4 DWORDs
  g1[1] = tile_d0 << 16;                        // tensor_dim0 [79:48]
  g1[2] = tile_d1 << 16;                        // tensor_dim1 [111:80] (lo 16)
  g1[3] = tile_d0 << 16;                        // tile_dim0   [127:112]
  g1[4] = tile_d1;                              // tile_dim1 [143:128], tile_dim2=0
  g1[5] = ten_d0_stride;                        // tensor_dim0_stride [207:160] (lo 32)
  g1[6] = 0u;                                   // stride hi, dim1_stride lo
  g1[7] = 0u;                                   // dim1_stride hi
  asm volatile("tensor_load_to_lds %0, %1"
               :: "s"(g0), "s"(g1)
               : "memory");
}

// ---- kernel 1: weight f32 -> bf16 ---------------------------------------

__global__ void cvt_w_kernel(const float* __restrict__ Wq, const float* __restrict__ Wk,
                             unsigned short* __restrict__ wq16, unsigned short* __restrict__ wk16) {
  int i = blockIdx.x * 256 + threadIdx.x;     // 128*256 = 32768 elements, 128 blocks
  wq16[i] = f2bf(Wq[i]);
  wk16[i] = f2bf(Wk[i]);
}

// ---- kernel 2: projection  out[b,n,o] = sum_c W[o,c]*x[b,c,n] + bias[o] --
// x: f32 [B, CI, NN];  w16: bf16 [CO, CI];  out: bf16 [B, NN, CO] (o contiguous)

__global__ __launch_bounds__(256) void proj_kernel(const float* __restrict__ x,
                                                   const unsigned short* __restrict__ w16,
                                                   const float* __restrict__ bias,
                                                   unsigned short* __restrict__ out) {
  __shared__ unsigned short xT[64 * 40];      // [n][c] transposed tile, ld = 40 halves (padded)

  const int tid  = threadIdx.x;
  const int wave = tid >> 5;                  // 8 waves: o-row tiles 16w..16w+15
  const int lane = tid & 31;
  const int ln   = lane & 15;
  const int hf   = lane >> 4;
  const int b    = blockIdx.y;
  const int nb   = blockIdx.x * 64;           // 64 spatial columns per block
  const float* xb = x + (size_t)b * CI * NN;

  v8f acc[4];
  #pragma unroll
  for (int ns = 0; ns < 4; ns++)
    #pragma unroll
    for (int i = 0; i < 8; i++) acc[ns][i] = 0.f;

  for (int cb = 0; cb < CI; cb += 32) {
    __syncthreads();
    // stage x[cb..cb+32) x [nb..nb+64) transposed into LDS as bf16 [n][c]
    #pragma unroll
    for (int i = 0; i < 8; i++) {
      int co = (tid >> 6) + i * 4;            // 0..31
      int n  = tid & 63;
      float v = xb[(size_t)(cb + co) * NN + nb + n];
      xT[n * 40 + co] = f2bf(v);
    }
    __syncthreads();

    // A fragment: 16 weight rows (o) x 32 (c), straight from global bf16
    v16bf af = frag_ld(w16 + (size_t)(16 * wave + ln) * CI + cb, hf * 8, 16 + hf * 8);
    #pragma unroll
    for (int ns = 0; ns < 4; ns++) {
      v16bf bf = frag_ld(&xT[(ns * 16 + ln) * 40], hf * 8, 16 + hf * 8);
      acc[ns] = wmma_bf16(af, bf, acc[ns]);
    }
  }

  // bias + transposed store to [n][o] layout
  #pragma unroll
  for (int i = 0; i < 8; i++) {
    int o = 16 * wave + i + hf * 8;           // C/D layout: VGPR i -> row i (+8 for hi half)
    float bv = bias[o];
    #pragma unroll
    for (int ns = 0; ns < 4; ns++) {
      int n = nb + ns * 16 + ln;
      out[((size_t)b * NN + n) * CO + o] = f2bf(acc[ns][i] + bv);
    }
  }
}

// ---- kernel 3: fused flash attention  out = softmax(Q Kt^T) V^T ----------
// v_src: f32 [B, CI, NN] (= b, the value tensor)
// Q, Kt: bf16 [B, NN, CO]
// out:   f32 [B, CI, NN]

__global__ __launch_bounds__(256) void attn_kernel(const float* __restrict__ v_src,
                                                   const unsigned short* __restrict__ Q,
                                                   const unsigned short* __restrict__ Kt,
                                                   float* __restrict__ out) {
  __shared__ unsigned short Kl[32 * 136];     // K tile [m][o], ld=136 halves
  __shared__ unsigned short Vl[256 * 40];     // V tile [c][m] bf16, ld=40 halves
  __shared__ unsigned short Pl[8 * 16 * 40];  // per-wave P tile [q][m], ld=40 halves

  const int tid  = threadIdx.x;
  const int wave = tid >> 5;
  const int lane = tid & 31;
  const int ln   = lane & 15;
  const int hf   = lane >> 4;
  const int b    = blockIdx.y;
  const int q0   = blockIdx.x * 128 + wave * 16;   // 16 query rows per wave

  const unsigned short* Qb  = Q  + (size_t)b * NN * CO;
  const unsigned short* Ktb = Kt + (size_t)b * NN * CO;
  const float*          Vb  = v_src + (size_t)b * CI * NN;
  unsigned short*       Pw  = Pl + wave * 16 * 40;

  // Q A-fragments (16 q-rows x 128 o), resident for the whole kernel
  v16bf qf[4];
  #pragma unroll
  for (int ch = 0; ch < 4; ch++)
    qf[ch] = frag_ld(Qb + (size_t)(q0 + ln) * CO + ch * 32, hf * 8, 16 + hf * 8);

  v8f acc[16];
  #pragma unroll
  for (int ct = 0; ct < 16; ct++)
    #pragma unroll
    for (int i = 0; i < 8; i++) acc[ct][i] = 0.f;

  float mrow[8], lrow[8];
  #pragma unroll
  for (int i = 0; i < 8; i++) { mrow[i] = -3.0e38f; lrow[i] = 0.f; }

  for (int mb = 0; mb < NN; mb += 32) {
    __syncthreads();
    // K tile rows 0..15: one TDM descriptor (wave 0), tracked on TENSORcnt
    if (wave == 0) {
      tdm_load_2d((unsigned)(uintptr_t)&Kl[0],
                  (unsigned long long)(uintptr_t)(Ktb + (size_t)mb * CO),
                  CO, 16, CO);
    }
    // K tile rows 16..31: per-lane async DMA (ASYNCcnt), 1 x b128 per thread
    {
      int row = 16 + (tid >> 4), c16 = tid & 15;
      async_g2l_b128(&Kl[row * 136 + c16 * 8],
                     Ktb + (size_t)(mb + row) * CO + c16 * 8);
    }
    // V tile: 256 rows x 32 f32, convert/pack to bf16 [c][m] (needs VALU cvt)
    #pragma unroll
    for (int i = 0; i < 8; i++) {
      int lin = tid + i * 256;                // 2048 x float4
      int c = lin >> 3, m4 = lin & 7;
      f32x4 v = *(const f32x4*)(Vb + (size_t)c * NN + mb + m4 * 4);
      u32x2 pk; pk[0] = pack2bf(v[0], v[1]); pk[1] = pack2bf(v[2], v[3]);
      *(u32x2*)&Vl[c * 40 + m4 * 4] = pk;
    }
    if (mb + 32 < NN) {                       // prefetch next tiles -> global_prefetch_b8
      __builtin_prefetch(Ktb + (size_t)(mb + 32 + (tid >> 3)) * CO, 0, 1);
      __builtin_prefetch(Vb + (size_t)tid * NN + mb + 32, 0, 1);
    }
    wait_async0();                            // s_wait_asynccnt 0
    if (wave == 0) __builtin_amdgcn_s_wait_tensorcnt(0);
    __syncthreads();

    // S = Q * K  (two 16x16 tiles covering 32 keys)
    v8f s0, s1;
    #pragma unroll
    for (int i = 0; i < 8; i++) { s0[i] = 0.f; s1[i] = 0.f; }
    #pragma unroll
    for (int ch = 0; ch < 4; ch++) {
      v16bf k0 = frag_ld(&Kl[ln * 136 + ch * 32],        hf * 8, 16 + hf * 8);
      v16bf k1 = frag_ld(&Kl[(16 + ln) * 136 + ch * 32], hf * 8, 16 + hf * 8);
      s0 = wmma_bf16(qf[ch], k0, s0);
      s1 = wmma_bf16(qf[ch], k1, s1);
    }

    // online softmax over the 32-key slab; rows live in 16-lane halves
    #pragma unroll
    for (int i = 0; i < 8; i++) {
      float t = fmaxf(s0[i], s1[i]);
      #pragma unroll
      for (int msk = 1; msk < 16; msk <<= 1) t = fmaxf(t, __shfl_xor(t, msk, 32));
      float mn = fmaxf(mrow[i], t);
      float sc = __expf(mrow[i] - mn);
      float p0 = __expf(s0[i] - mn);
      float p1 = __expf(s1[i] - mn);
      float rs = p0 + p1;
      #pragma unroll
      for (int msk = 1; msk < 16; msk <<= 1) rs += __shfl_xor(rs, msk, 32);
      lrow[i] = lrow[i] * sc + rs;
      mrow[i] = mn;
      int prow = i + hf * 8;
      Pw[prow * 40 + ln]      = f2bf(p0);
      Pw[prow * 40 + 16 + ln] = f2bf(p1);
      #pragma unroll
      for (int ct = 0; ct < 16; ct++) acc[ct][i] *= sc;
    }

    // P * V : convert P via wave-private LDS into A-fragment layout
    v16bf pf = frag_ld(&Pw[ln * 40], hf * 8, 16 + hf * 8);
    #pragma unroll
    for (int ct = 0; ct < 16; ct++) {
      v16bf vf = frag_ld(&Vl[(ct * 16 + ln) * 40], hf * 8, 16 + hf * 8);
      acc[ct] = wmma_bf16(pf, vf, acc[ct]);
    }
  }

  // epilogue: normalize by row-sum, store out[b, c, n]
  #pragma unroll
  for (int ct = 0; ct < 16; ct++) {
    int c = ct * 16 + ln;
    float* ob = out + ((size_t)b * CI + c) * NN;
    #pragma unroll
    for (int i = 0; i < 8; i++) {
      ob[q0 + i + hf * 8] = acc[ct][i] / lrow[i];
    }
  }
}

// ---- launcher ------------------------------------------------------------

extern "C" void kernel_launch(void* const* d_in, const int* in_sizes, int n_in,
                              void* d_out, int out_size, void* d_ws, size_t ws_size,
                              hipStream_t stream) {
  const float* p  = (const float*)d_in[0];
  const float* bb = (const float*)d_in[1];
  const float* Wq = (const float*)d_in[2];
  const float* bq = (const float*)d_in[3];
  const float* Wk = (const float*)d_in[4];
  const float* bk = (const float*)d_in[5];
  float* out = (float*)d_out;

  unsigned char* ws = (unsigned char*)d_ws;
  unsigned short* wq16 = (unsigned short*)(ws);                       // 64 KB
  unsigned short* wk16 = (unsigned short*)(ws + 65536);               // 64 KB
  unsigned short* Qws  = (unsigned short*)(ws + 131072);              // 8.4 MB  [B,N,CO] bf16
  unsigned short* Ktws = (unsigned short*)(ws + 131072 + (size_t)B_ * NN * CO * 2);

  cvt_w_kernel<<<128, 256, 0, stream>>>(Wq, Wk, wq16, wk16);

  dim3 pg(NN / 64, B_);
  proj_kernel<<<pg, 256, 0, stream>>>(p,  wq16, bq, Qws);   // Q  [B,N,128]
  proj_kernel<<<pg, 256, 0, stream>>>(bb, wk16, bk, Ktws);  // Kt [B,N,128]

  dim3 ag(NN / 128, B_);
  attn_kernel<<<ag, 256, 0, stream>>>(bb, Qws, Ktws, out);
}
